// GNN_26852135535008
// MI455X (gfx1250) — matrix operand order repaired and checked by reading, not compiled, not run on hardware
//
#include <hip/hip_runtime.h>
#include <hip/hip_bf16.h>
#include <stdint.h>

// ---------------------------------------------------------------------------
// GIN (3-layer) for MI455X / gfx1250, wave32, WMMA f32_16x16x32_f16
//   layer: h = relu((x + segment_sum(x[src], dst)) @ W + b)
// N=50000 nodes, E=1.6M edges, F=128, classes=2
//
// Aggregation is the runtime dominator (3 x 204M f32 atomics, all L2-resident:
// feature table = 25.6MB << 192MB L2). GEMMs (1.64 GFLOP each) run on WMMA
// with pre-swizzled f16 fragments so the inner loop is pure b128 loads + wmma.
// ---------------------------------------------------------------------------

#define N_FEAT 128

typedef __attribute__((ext_vector_type(16))) _Float16 v16h;
typedef __attribute__((ext_vector_type(4)))  _Float16 v4h;
typedef __attribute__((ext_vector_type(8)))  float    v8f;

// ---------------------------------------------------------------------------
// Zero a float buffer (float4 granularity; sizes here are multiples of 4).
// ---------------------------------------------------------------------------
__global__ __launch_bounds__(256) void gin_zero_f32(float* __restrict__ p, long long n4) {
    long long i = (long long)blockIdx.x * blockDim.x + threadIdx.x;
    if (i < n4) {
        float4 z = {0.f, 0.f, 0.f, 0.f};
        ((float4*)p)[i] = z;
    }
}

// ---------------------------------------------------------------------------
// Swizzle f32 weights W[k][col] (128x128 row-major) into WMMA B-fragment
// order as f16:  Wswz[((colTile*4 + kcIdx)*32 + lane)*16 + e]
// where  K = kcIdx*32 + (e/8)*16 + (lane/16)*8 + e%8,  col = colTile*16 + (lane%16).
// Compute kernel then reads one contiguous 32B v16h per lane per fragment.
// ---------------------------------------------------------------------------
__global__ __launch_bounds__(256) void gin_swizzle_w(const float* __restrict__ W,
                                                     _Float16* __restrict__ Wswz) {
    int t = blockIdx.x * blockDim.x + threadIdx.x;   // [0, 8*4*32*16 = 16384)
    int e       = t & 15;
    int lane    = (t >> 4) & 31;
    int kcIdx   = (t >> 9) & 3;
    int colTile = t >> 11;
    int K   = kcIdx * 32 + ((e >> 3) << 4) + ((lane >> 4) << 3) + (e & 7);
    int col = colTile * 16 + (lane & 15);
    Wswz[t] = (_Float16)W[(size_t)K * N_FEAT + col];
}

// ---------------------------------------------------------------------------
// Scatter-add: agg[dst[e]][:] += X[src[e]][:]
// One thread per (edge, 4-feature chunk): float4 gather + 4 f32 atomics.
// ---------------------------------------------------------------------------
__global__ __launch_bounds__(256) void gin_scatter_add(const float* __restrict__ X,
                                                       const int* __restrict__ srcI,
                                                       const int* __restrict__ dstI,
                                                       float* __restrict__ AGG,
                                                       int nedges) {
    long long gid  = (long long)blockIdx.x * blockDim.x + threadIdx.x;
    int       edge = (int)(gid >> 5);          // 32 chunks of 4 floats per edge
    int       c    = ((int)gid & 31) * 4;
    if (edge >= nedges) return;
    int s = srcI[edge];
    int d = dstI[edge];
    const float4 v = *(const float4*)(X + (size_t)s * N_FEAT + c);
    float* p = AGG + (size_t)d * N_FEAT + c;
    atomicAdd(p + 0, v.x);
    atomicAdd(p + 1, v.y);
    atomicAdd(p + 2, v.z);
    atomicAdd(p + 3, v.w);
}

// ---------------------------------------------------------------------------
// Fused GIN layer: OUT = relu((X + AGG) @ W + bias), 128 -> 128.
// Block = 256 threads = 8 waves; block computes 16 rows x 128 cols.
// Wave w owns col-tile w (16 cols); K=128 -> 4x v_wmma_f32_16x16x32_f16.
//
// H-tile is staged into LDS as f16 ALREADY in A-fragment order:
//   A layout (16-bit A 16x32): K%32 = g*16 + h*8 + w  ->  lane = h*16 + row,
//   elem e = g*8 + w. A float4 (4 consecutive k, 8-aligned-within-block) maps
//   to 4 contiguous halves -> single 8B LDS store. Compute reads one
//   contiguous 32B v16h per lane per k-chunk (2x ds_load_b128).
// ---------------------------------------------------------------------------
__global__ __launch_bounds__(256) void gin_gemm_relu(const float* __restrict__ X,
                                                     const float* __restrict__ AGG,
                                                     const _Float16* __restrict__ Wswz,
                                                     const float* __restrict__ bias,
                                                     float* __restrict__ OUT) {
    // Sh[kcIdx][lane][e] : 4*32*16 halves = 4KB
    __shared__ __align__(32) _Float16 Sh[4 * 32 * 16];

    const int tid     = threadIdx.x;
    const int lane    = tid & 31;               // wave32
    const int wave    = tid >> 5;               // 0..7 -> col tile
    const int rowBase = blockIdx.x * 16;        // 50000 = 3125 * 16, exact

    // Stage H = X + AGG for 16 rows: 512 float4s, 2 per thread, converted to
    // f16 and scattered into A-fragment order (4 contiguous halves per store).
    for (int i = tid; i < 16 * N_FEAT / 4; i += 256) {
        int r  = i >> 5;                        // 32 float4 per row
        int c4 = i & 31;
        size_t g = (size_t)(rowBase + r) * N_FEAT;
        const float4 xv = ((const float4*)(X + g))[c4];
        const float4 av = ((const float4*)(AGG + g))[c4];
        int k     = c4 * 4;
        int kcIdx = k >> 5;
        int kr    = k & 31;
        int lane_s = (((kr >> 3) & 1) << 4) + r;         // h*16 + row
        int e0     = ((kr >> 4) << 3) + (kr & 7);        // g*8 + w
        v4h hv = {(_Float16)(xv.x + av.x), (_Float16)(xv.y + av.y),
                  (_Float16)(xv.z + av.z), (_Float16)(xv.w + av.w)};
        *(v4h*)(Sh + ((kcIdx * 32 + lane_s) << 4) + e0) = hv;
    }
    __syncthreads();

    v8f acc = {};
    const _Float16* wp = Wswz + (((size_t)wave * 4) << 9) + (lane << 4);

    #pragma unroll
    for (int kcIdx = 0; kcIdx < 4; ++kcIdx) {
        v16h a = *(const v16h*)(Sh + ((kcIdx * 32 + lane) << 4));   // 32B LDS read
        v16h b = *(const v16h*)(wp + (kcIdx << 9));                 // 32B coalesced
        acc = __builtin_amdgcn_wmma_f32_16x16x32_f16(
            /*neg_a=*/false, a, /*neg_b=*/false, b,
            /*c_mod=*/(short)0, acc, /*reuse_a=*/false, /*reuse_b=*/false);
    }

    // D epilogue: VGPR r -> row = r + 8*(lane/16), col = lane%16 of the tile.
    const int hi  = lane >> 4;
    const int col = wave * 16 + (lane & 15);
    const float bv = bias[col];
    #pragma unroll
    for (int r = 0; r < 8; ++r) {
        int row = rowBase + r + (hi << 3);
        float v = acc[r] + bv;
        OUT[(size_t)row * N_FEAT + col] = v > 0.f ? v : 0.f;
    }
}

// ---------------------------------------------------------------------------
// Final layer: OUT[n][0:2] = (H + AGG)[n] @ W3 + b3.  N-dim is 2 << 16,
// so WMMA would waste 87% of the tile; plain VALU dot products instead.
// ---------------------------------------------------------------------------
__global__ __launch_bounds__(128) void gin_final(const float* __restrict__ H,
                                                 const float* __restrict__ AGG,
                                                 const float* __restrict__ W3,
                                                 const float* __restrict__ b3,
                                                 float* __restrict__ OUT, int nrows) {
    int row = blockIdx.x * blockDim.x + threadIdx.x;
    if (row >= nrows) return;
    float acc0 = b3[0], acc1 = b3[1];
    const float* h = H   + (size_t)row * N_FEAT;
    const float* a = AGG + (size_t)row * N_FEAT;
    #pragma unroll 8
    for (int k = 0; k < N_FEAT; ++k) {
        float s = h[k] + a[k];
        acc0 = fmaf(s, W3[k * 2 + 0], acc0);
        acc1 = fmaf(s, W3[k * 2 + 1], acc1);
    }
    OUT[(size_t)row * 2 + 0] = acc0;
    OUT[(size_t)row * 2 + 1] = acc1;
}

// ---------------------------------------------------------------------------
// Host-side orchestration (graph-capture safe: only kernel launches).
// ---------------------------------------------------------------------------
extern "C" void kernel_launch(void* const* d_in, const int* in_sizes, int n_in,
                              void* d_out, int out_size, void* d_ws, size_t ws_size,
                              hipStream_t stream) {
    const float* x  = (const float*)d_in[0];
    const int*   ei = (const int*)d_in[1];
    const float* W1 = (const float*)d_in[2];
    const float* b1 = (const float*)d_in[3];
    const float* W2 = (const float*)d_in[4];
    const float* b2 = (const float*)d_in[5];
    const float* W3 = (const float*)d_in[6];
    const float* b3 = (const float*)d_in[7];
    float*       outp = (float*)d_out;

    const int nedges = in_sizes[1] / 2;             // edge_index is [2, E]
    const int nnodes = in_sizes[0] / N_FEAT;        // 50000
    const int* srcI = ei;
    const int* dstI = ei + nedges;

    // Workspace layout: agg | hA | hB | W1swz | W2swz  (~77 MB)
    size_t nf = (size_t)nnodes * N_FEAT;
    float*    agg = (float*)d_ws;
    float*    hA  = agg + nf;
    float*    hB  = hA + nf;
    _Float16* W1s = (_Float16*)(hB + nf);
    _Float16* W2s = W1s + N_FEAT * N_FEAT;

    // Weight swizzle+convert to f16 fragments (small; each call, deterministic).
    {
        int n = N_FEAT * N_FEAT;                     // 16384
        gin_swizzle_w<<<n / 256, 256, 0, stream>>>(W1, W1s);
        gin_swizzle_w<<<n / 256, 256, 0, stream>>>(W2, W2s);
    }

    long long n4 = (long long)nf / 4;
    int zgrid = (int)((n4 + 255) / 256);
    long long sthreads = (long long)nedges * 32;
    int sgrid = (int)((sthreads + 255) / 256);
    dim3 ggrid(nnodes / 16);                        // 3125 blocks, 8 waves each
    dim3 gblock(256);

    // Layer 1: agg = scatter(x); hA = relu((x+agg)@W1 + b1)
    gin_zero_f32<<<zgrid, 256, 0, stream>>>(agg, n4);
    gin_scatter_add<<<sgrid, 256, 0, stream>>>(x, srcI, dstI, agg, nedges);
    gin_gemm_relu<<<ggrid, gblock, 0, stream>>>(x, agg, W1s, b1, hA);

    // Layer 2
    gin_zero_f32<<<zgrid, 256, 0, stream>>>(agg, n4);
    gin_scatter_add<<<sgrid, 256, 0, stream>>>(hA, srcI, dstI, agg, nedges);
    gin_gemm_relu<<<ggrid, gblock, 0, stream>>>(hA, agg, W2s, b2, hB);

    // Layer 3 (128 -> 2, no relu)
    gin_zero_f32<<<zgrid, 256, 0, stream>>>(agg, n4);
    gin_scatter_add<<<sgrid, 256, 0, stream>>>(hB, srcI, dstI, agg, nedges);
    gin_final<<<(nnodes + 127) / 128, 128, 0, stream>>>(hB, agg, W3, b3, outp, nnodes);
}
